// OrbitalTransformer_42374147342462
// MI455X (gfx1250) — matrix operand order repaired and checked by reading, not compiled, not run on hardware
//
#include <hip/hip_runtime.h>
#include <math.h>

typedef __attribute__((ext_vector_type(16))) _Float16 v16h;
typedef __attribute__((ext_vector_type(8)))  _Float16 v8h;
typedef __attribute__((ext_vector_type(8)))  float    v8f;
typedef __attribute__((ext_vector_type(4)))  unsigned int u32x4;
typedef __attribute__((ext_vector_type(8)))  int      i32x8;
typedef __attribute__((ext_vector_type(4)))  int      i32x4;

#define B_   16
#define N_   16
#define O_   64
#define H_   4
#define NROW (B_*N_*O_)     // 16384
#define DINP 224            // padded MLP input width (both iterations)

#if defined(__has_builtin)
#  if __has_builtin(__builtin_amdgcn_tensor_load_to_lds)
#    define HAVE_TDM 1
#  endif
#endif
#ifndef HAVE_TDM
#  define HAVE_TDM 0
#endif

// ---------------------------------------------------------------------------
// Tensor Data Mover: 1-row 2-byte-element tile, contiguous copy global->LDS.
// Descriptor per CDNA5 ISA ch.8 (group0: count|lds|global|type, group1: dsize,
// dims). nhalves must fit tile_dim0 (16 bits).
// This toolchain exposes the 6-arg builtin:
//   (u32x4 g0, i32x8 g1, i32x4 g2, i32x4 g3, i32x8 extra, i32 cpol)
// ---------------------------------------------------------------------------
#if HAVE_TDM
static __device__ __forceinline__ void tdm_copy_to_lds(const _Float16* gsrc,
                                                       void* lds_dst, int nhalves) {
  unsigned lds = (unsigned)(size_t)lds_dst;          // low 32b of generic = LDS offset
  unsigned long long ga = (unsigned long long)(size_t)gsrc;
  u32x4 g0;
  g0[0] = 1u;                                        // count=1 (valid user D#)
  g0[1] = lds;                                       // lds_addr
  g0[2] = (unsigned)(ga & 0xffffffffu);              // global_addr[31:0]
  g0[3] = (unsigned)((ga >> 32) & 0x1ffffffu) | (2u << 30);  // addr[56:32] | type=2
  i32x8 g1;
  g1[0] = (1 << 16);                                 // data_size: 1 -> 2 bytes
  g1[1] = (nhalves & 0xffff) << 16;                  // tensor_dim0[15:0] @bits63:48
  g1[2] = ((nhalves >> 16) & 0xffff) | (1 << 16);    // tensor_dim0[31:16] | tensor_dim1=1
  g1[3] = (nhalves & 0xffff) << 16;                  // tile_dim0 @bits127:112
  g1[4] = 1;                                         // tile_dim1=1, tile_dim2=0
  g1[5] = nhalves;                                   // tensor_dim0_stride[31:0]
  g1[6] = (nhalves & 0xffff) << 16;                  // dim0_str[47:32]=0 | dim1_str[15:0]
  g1[7] = 0;
  i32x4 z4 = {0, 0, 0, 0};
  i32x8 z8 = {0, 0, 0, 0, 0, 0, 0, 0};
  __builtin_amdgcn_tensor_load_to_lds(g0, g1, z4, z4, z8, 0);
}
#endif

static __device__ __forceinline__ void stage_weights(const _Float16* gsrc,
                                                     _Float16* lds_dst, int nhalves) {
#if HAVE_TDM
  if (threadIdx.x < 32) {
    tdm_copy_to_lds(gsrc, lds_dst, nhalves);
  }
#else
  for (int idx = threadIdx.x; idx < nhalves; idx += blockDim.x)
    lds_dst[idx] = gsrc[idx];
#endif
}
static __device__ __forceinline__ void stage_weights_wait() {
#if HAVE_TDM
  if (threadIdx.x < 32) __builtin_amdgcn_s_wait_tensorcnt(0);
#endif
}

// ---------------------------------------------------------------------------
// WMMA f16 16x16x32 helpers (gfx1250 wave32, ISA 7.12.2). Operands are stored
// K-contiguous so each lane's 16 halves are two aligned 16-byte runs:
//   A (row-major M x K):  lane(m,hi): K = hi*8+[0..8) and 16+hi*8+[0..8)
//   B (N-major  N x K):   lane(n,hi): K = hi*8+[0..8) and 16+hi*8+[0..8)
//   D f32: reg r, lane -> (M = r + 8*hi, N = lane&15)
// ---------------------------------------------------------------------------
static __device__ __forceinline__ v16h ld_a(const _Float16* p, int ld, int m0, int k0) {
  int lane = threadIdx.x & 31;
  int m = m0 + (lane & 15);
  int hi = lane >> 4;
  const _Float16* q = p + m*ld + k0 + hi*8;
  v8h lo = *(const v8h*)q;
  v8h hh = *(const v8h*)(q + 16);
  v16h a;
#pragma unroll
  for (int j = 0; j < 8; ++j) { a[j] = lo[j]; a[j + 8] = hh[j]; }
  return a;
}
static __device__ __forceinline__ v16h ld_bT(const _Float16* p, int ld, int k0, int n0) {
  int lane = threadIdx.x & 31;
  int n = n0 + (lane & 15);
  int hi = lane >> 4;
  const _Float16* q = p + n*ld + k0 + hi*8;
  v8h lo = *(const v8h*)q;
  v8h hh = *(const v8h*)(q + 16);
  v16h b;
#pragma unroll
  for (int j = 0; j < 8; ++j) { b[j] = lo[j]; b[j + 8] = hh[j]; }
  return b;
}
// C[m,n] += sum_k A[m0+m, k] * B[n0+n, k]   (A row-major MxK, B N-major NxK)
static __device__ __forceinline__ v8f mmacc(const _Float16* A, int lda,
                                            const _Float16* Bt, int ldb,
                                            int m0, int n0, int K, v8f acc) {
#pragma unroll
  for (int k0 = 0; k0 < K; k0 += 32) {
    v16h a = ld_a(A, lda, m0, k0);
    v16h b = ld_bT(Bt, ldb, k0, n0);
    acc = __builtin_amdgcn_wmma_f32_16x16x32_f16(false, a, false, b, (short)0, acc,
                                                 false, false);
  }
  return acc;
}

// ---------------------------------------------------------------------------
// Kernel 1: build padded state s(.,64), v(.,32,3), RBF edge_attr, Y1
// ---------------------------------------------------------------------------
__global__ void k_prep_state(const float* __restrict__ scal, const float* __restrict__ vec,
                             const float* __restrict__ diff, const float* __restrict__ dist,
                             const int* __restrict__ noccp,
                             float* __restrict__ s, float* __restrict__ v,
                             float* __restrict__ ea, float* __restrict__ y1) {
  const int SE = NROW*64;
  const int VE = NROW*32*3;
  const int EE = B_*N_*N_*32;
  const int YE = B_*N_*N_*3;
  int t = blockIdx.x*256 + threadIdx.x;
  int nocc = *noccp;
  if (t < SE) {
    int row = t / 64, c = t % 64;
    int o = row % O_;
    float val = 0.f;
    if (c < 16) val = scal[row*16 + c];
    else if (c < 33) {
      int pc = c - 16;
      float pos = (float)(o - nocc) / (float)O_;
      if (pc < 8)       val = __sinf(pos * 3.14159265358979f * (float)(pc + 1));
      else if (pc < 16) val = __cosf(pos * 3.14159265358979f * (float)(pc - 7));
      else              val = (o < nocc) ? 1.f : -1.f;
    }
    s[t] = val;
  } else if ((t -= SE) < VE) {
    int row = t / 96, r = t % 96;
    int ch = r / 3, x = r % 3;
    v[t] = (ch < 8) ? vec[(row*8 + ch)*3 + x] : 0.f;
  } else if ((t -= VE) < EE) {
    int e = t / 32, r = t % 32;
    float c = 5.0f * (float)r / 31.0f;
    float d = (dist[e] - c) * (32.0f / 5.0f);
    ea[t] = __expf(-0.5f * d * d);
  } else if ((t -= EE) < YE) {
    y1[t] = 1.7320508075688772f * diff[t];
  }
}

// ---------------------------------------------------------------------------
// Kernel 2: weights -> f16, zero-padded, fan-in scale folded, TRANSPOSED
// (N-major, K-contiguous) for vectorized WMMA B-operand loads.
// wf layout (halves):
//  attn a in {ion0,ion1,orb0,orb1}: a*15360 +
//    [Wq_sT 64x64 | Wk_sT 64x64 | Wv_sT 64x64 | Wq_v 32x32 | Wk_v | Wv_v]
//  mlp m in {0,1}: 61440 + m*19456 + [W1pT 64x224 | W2pT 64x64 | W3pT 16x64]
// ---------------------------------------------------------------------------
struct PPtrs { const float* p[36]; };

__global__ void k_prep_params(PPtrs P, _Float16* __restrict__ wf) {
  const int ATTN_H = 15360, MLP_H = 19456;
  const int NATTN = 4*ATTN_H, TOT = NATTN + 2*MLP_H;
  int tid = blockIdx.x*256 + threadIdx.x;
  if (tid >= TOT) return;
  float val = 0.f;
  if (tid < NATTN) {
    int a = tid / ATTN_H, r = tid % ATTN_H;
    int pb   = (a == 0) ? 0 : (a == 1) ? 12 : (a == 2) ? 24 : 30;
    int sin_ = (a == 0) ? 33 : 64;
    int vin_ = (a == 0) ? 8 : 32;
    if (r < 12288) {                       // transposed scalar weights
      int which = r / 4096, idx = r % 4096;
      int c = idx / 64, k = idx % 64;
      int rel = (which == 0) ? 2 : (which == 1) ? 0 : 4;   // Wq_s, Wk_s, Wv_s
      if (k < sin_) val = P.p[pb + rel][k*64 + c] * rsqrtf((float)sin_);
    } else {                               // vector weights, kept [k][c]
      int rv = r - 12288;
      int which = rv / 1024, idx = rv % 1024;
      int k = idx / 32, c = idx % 32;
      int rel = (which == 0) ? 3 : (which == 1) ? 1 : 5;   // Wq_v, Wk_v, Wv_v
      if (k < vin_) val = P.p[pb + rel][k*32 + c] * rsqrtf((float)vin_);
    }
  } else {
    int t = tid - NATTN;
    int m = t / MLP_H, r = t % MLP_H;
    int pb = (m == 0) ? 6 : 18;
    int sin_ = (m == 0) ? 33 : 64, vin_ = (m == 0) ? 8 : 32;
    if (r < 14336) {                       // W1pT: [c][k], k = padded feature idx
      int c = r / DINP, k = r % DINP;
      int orow = -1;
      if (k < 32) orow = k;
      else if (k < 96)  { int kk = k - 32;  if (kk < sin_) orow = 32 + kk; }
      else if (k < 128) { int kk = k - 96;  if (kk < vin_) orow = 32 + sin_ + kk; }
      else if (k < 192) { int kk = k - 128; if (kk < sin_) orow = 32 + sin_ + vin_ + kk; }
      else              { int kk = k - 192; if (kk < vin_) orow = 32 + 2*sin_ + vin_ + kk; }
      if (orow >= 0) val = P.p[pb + 0][orow*64 + c];
    } else if (r < 18432) {                // W2pT [c][k]
      int idx = r - 14336;
      int c = idx / 64, k = idx % 64;
      val = P.p[pb + 1][k*64 + c];
    } else {                               // W3pT [c(16)][k(64)]
      int idx = r - 18432;
      int c = idx / 64, k = idx % 64;
      if (c < 4) val = P.p[pb + 2][k*4 + c];
    }
  }
  wf[tid] = (_Float16)val;
}

// ---------------------------------------------------------------------------
// Kernel 3: scalar q/k/v projection GEMM (M=16384, K=64, N=64), z = q/k/v.
// Weights are TDM-staged into LDS while activations are converted to f16.
// ---------------------------------------------------------------------------
__global__ void k_proj_s(const float* __restrict__ S, const _Float16* __restrict__ Wbase,
                         float* __restrict__ Out) {
  __shared__ __align__(16) _Float16 sh[64*64];
  __shared__ __align__(16) _Float16 shW[64*64];
  int z = blockIdx.z;
  const _Float16* W = Wbase + z*4096;      // transposed [c][k]
  float* out = Out + (size_t)z * NROW * 64;
  int row0 = blockIdx.x * 64;
  stage_weights(W, shW, 4096);
  for (int idx = threadIdx.x; idx < 64*64; idx += 128)
    sh[idx] = (_Float16)S[(size_t)(row0 + (idx >> 6))*64 + (idx & 63)];
  stage_weights_wait();
  __syncthreads();
  int wv = threadIdx.x >> 5, lane = threadIdx.x & 31;
  int m0 = wv * 16, n = lane & 15, hi = lane >> 4;
  for (int nt = 0; nt < 4; ++nt) {
    v8f acc = {};
    acc = mmacc(sh, 64, shW, 64, m0, nt*16, 64, acc);
#pragma unroll
    for (int r = 0; r < 8; ++r)
      out[(size_t)(row0 + m0 + r + 8*hi)*64 + nt*16 + n] = acc[r];
  }
}

// ---------------------------------------------------------------------------
// Kernel 4: vector q/k/v projection (tiny contraction over 32 padded channels)
// ---------------------------------------------------------------------------
__global__ void k_proj_v(const float* __restrict__ V, const _Float16* __restrict__ Wbase,
                         float* __restrict__ Out) {
  int z = blockIdx.z;
  const _Float16* W = Wbase + 12288 + z*1024;
  float* out = Out + (size_t)z * NROW * 96;
  int tid = blockIdx.x*256 + threadIdx.x;
  if (tid >= NROW*96) return;
  int row = tid / 96, r = tid % 96;
  int w = r / 3, x = r % 3;
  float acc = 0.f;
#pragma unroll
  for (int k = 0; k < 32; ++k)
    acc += V[((size_t)row*32 + k)*3 + x] * (float)W[k*32 + w];
  out[tid] = acc;
}

// ---------------------------------------------------------------------------
// Kernel 5: edge MLP. Block per (b,i,j): 64x224 f16 features built in LDS,
// 224->64 tanh -> 64->64 tanh -> 64->4, all WMMA. W2/W3 TDM-staged, W1
// prefetched and streamed from L2.
// ---------------------------------------------------------------------------
__global__ void k_edge_mlp(const float* __restrict__ S, const float* __restrict__ V,
                           const float* __restrict__ EA, const float* __restrict__ Y1,
                           const _Float16* __restrict__ Wm,
                           const float* __restrict__ b1, const float* __restrict__ b2,
                           const float* __restrict__ b3,
                           float* __restrict__ EW) {
  __shared__ __align__(16) _Float16 feat[64*DINP];
  __shared__ __align__(16) _Float16 h1[64*64];
  __shared__ __align__(16) _Float16 h2[64*64];
  __shared__ __align__(16) _Float16 shW23[5120];     // W2pT (4096) | W3pT (1024)
  int e = blockIdx.x;                  // b*256 + i*16 + j
  int b = e >> 8, i = (e >> 4) & 15, j = e & 15;
  stage_weights(Wm + 14336, shW23, 5120);
  if (threadIdx.x < 64)
    __builtin_prefetch(Wm + threadIdx.x*DINP, 0, 3);   // global_prefetch W1pT
  const float* y = Y1 + (size_t)e*3;
  float y0 = y[0], y1v = y[1], y2 = y[2];
  int rowI = (b*16 + i)*64;
  int rowJ = (b*16 + j)*64;
  const float inv_s3 = 0.57735026918962576f;
  for (int idx = threadIdx.x; idx < 64*DINP; idx += 128) {
    int o = idx / DINP, c = idx % DINP;
    float val;
    if (c < 32)       val = EA[(size_t)e*32 + c];
    else if (c < 96)  val = S[(size_t)(rowI + o)*64 + (c - 32)];
    else if (c < 128) {
      const float* vp = V + ((size_t)(rowI + o)*32 + (c - 96))*3;
      val = (vp[0]*y0 + vp[1]*y1v + vp[2]*y2) * inv_s3;
    }
    else if (c < 192) val = S[(size_t)(rowJ + o)*64 + (c - 128)];
    else {
      const float* vp = V + ((size_t)(rowJ + o)*32 + (c - 192))*3;
      val = (vp[0]*y0 + vp[1]*y1v + vp[2]*y2) * inv_s3;
    }
    feat[idx] = (_Float16)val;
  }
  stage_weights_wait();
  __syncthreads();
  const _Float16* W1 = Wm;                 // [c(64)][k(224)]
  const _Float16* W2 = shW23;              // [c(64)][k(64)]
  const _Float16* W3 = shW23 + 4096;       // [c(16)][k(64)]
  int wv = threadIdx.x >> 5, lane = threadIdx.x & 31;
  int m0 = wv*16, n = lane & 15, hi = lane >> 4;
  for (int nt = 0; nt < 4; ++nt) {
    v8f acc = {};
    acc = mmacc(feat, DINP, W1, DINP, m0, nt*16, DINP, acc);
    float bb = b1[nt*16 + n];
#pragma unroll
    for (int r = 0; r < 8; ++r)
      h1[(m0 + r + 8*hi)*64 + nt*16 + n] = (_Float16)tanhf(acc[r] + bb);
  }
  __syncthreads();
  for (int nt = 0; nt < 4; ++nt) {
    v8f acc = {};
    acc = mmacc(h1, 64, W2, 64, m0, nt*16, 64, acc);
    float bb = b2[nt*16 + n];
#pragma unroll
    for (int r = 0; r < 8; ++r)
      h2[(m0 + r + 8*hi)*64 + nt*16 + n] = (_Float16)tanhf(acc[r] + bb);
  }
  __syncthreads();
  {
    v8f acc = {};
    acc = mmacc(h2, 64, W3, 64, m0, 0, 64, acc);
    if (n < 4) {
      float bb = b3[n];
#pragma unroll
      for (int r = 0; r < 8; ++r) {
        int o = m0 + r + 8*hi;
        EW[((size_t)e*64 + o)*4 + n] = acc[r] + bb;
      }
    }
  }
}

// ---------------------------------------------------------------------------
// Kernel 6: ion attention. Block per (b,o); wave = head. Transposed logits via
// WMMA (softmax dim in the register dim), shfl_xor(16) reduction, WMMA
// aggregation with V staged N-major (K-contiguous).
// ---------------------------------------------------------------------------
__global__ void k_ion_attn(const float* __restrict__ QS, const float* __restrict__ KS,
                           const float* __restrict__ VSC,
                           const float* __restrict__ QV, const float* __restrict__ KV,
                           const float* __restrict__ VVE,
                           const float* __restrict__ EW,
                           const float* __restrict__ Sin, const float* __restrict__ Vin,
                           float* __restrict__ Sout, float* __restrict__ Vout,
                           int resid) {
  __shared__ __align__(16) _Float16 qa[4][16*64];   // [i][k]
  __shared__ __align__(16) _Float16 ka[4][16*64];   // [j][k]
  __shared__ __align__(16) _Float16 va[4][48*32];   // [c][j] (K-contiguous, j pad 32)
  __shared__ __align__(16) _Float16 wt[4][16*32];   // [i][j] (j pad 32)
  int b = blockIdx.x >> 6;
  int o = blockIdx.x & 63;
  int h = threadIdx.x >> 5;
  int lane = threadIdx.x & 31;
  const float ANORM  = 0.20412414523193154f;                 // 1/sqrt(24)
  const float ANORM3 = ANORM * 0.57735026918962576f;         // folded 1/sqrt(3)
  _Float16* q  = qa[h];
  _Float16* kk = ka[h];
  _Float16* vt = va[h];
  _Float16* w_ = wt[h];
  for (int idx = lane; idx < 16*64; idx += 32) {
    int i = idx >> 6, c = idx & 63;
    size_t row = (size_t)((b*16 + i)*64 + o);
    float fq, fk;
    if (c < 16) {
      size_t oi = row*64 + h*16 + c;
      fq = QS[oi] * ANORM; fk = KS[oi];
    } else if (c < 40) {
      int cc = c - 16;
      size_t oi = row*96 + (h*8 + cc/3)*3 + cc%3;
      fq = QV[oi] * ANORM3; fk = KV[oi];
    } else { fq = fk = 0.f; }
    q[idx]  = (_Float16)fq;
    kk[idx] = (_Float16)fk;
  }
  for (int idx = lane; idx < 48*16; idx += 32) {     // V, N-major [c][j]
    int c = idx >> 4, j = idx & 15;
    size_t row = (size_t)((b*16 + j)*64 + o);
    float fv;
    if (c < 16)      fv = VSC[row*64 + h*16 + c];
    else if (c < 40) { int cc = c - 16; fv = VVE[row*96 + (h*8 + cc/3)*3 + cc%3]; }
    else             fv = 0.f;
    vt[c*32 + j]      = (_Float16)fv;
    vt[c*32 + 16 + j] = (_Float16)0.f;               // zero K-pad
  }
  for (int idx = lane; idx < 16*16; idx += 32)       // zero w K-pad
    w_[(idx >> 4)*32 + 16 + (idx & 15)] = (_Float16)0.f;
  __syncthreads();
  // logits^T: D(m=j, n=i) = sum_k K[j,k] * Q[i,k]
  v8f lt = {};
  lt = mmacc(kk, 64, q, 64, 0, 0, 64, lt);
  int i = lane & 15, hi = lane >> 4;
  float t[8], mx = -1e30f;
#pragma unroll
  for (int r = 0; r < 8; ++r) {
    int j = r + 8*hi;
    t[r] = lt[r] + EW[((((size_t)b*16 + i)*16 + j)*64 + o)*4 + h];
    mx = fmaxf(mx, t[r]);
  }
  mx = fmaxf(mx, __shfl_xor(mx, 16));
  float sum = 0.f;
#pragma unroll
  for (int r = 0; r < 8; ++r) { t[r] = __expf(t[r] - mx); sum += t[r]; }
  sum += __shfl_xor(sum, 16);
  float inv = 1.f / sum;
#pragma unroll
  for (int r = 0; r < 8; ++r)
    w_[i*32 + 8*hi + r] = (_Float16)(t[r] * inv);    // row-major [i][j], contiguous
  __syncthreads();
  // aggregation: D(m=i, n=c) = sum_j w[i,j] * V[c,j]
  for (int nt = 0; nt < 3; ++nt) {
    v8f acc = {};
    acc = mmacc(w_, 32, vt, 32, 0, nt*16, 32, acc);
    int n = lane & 15;
#pragma unroll
    for (int r = 0; r < 8; ++r) {
      int m = r + 8*hi;
      size_t row = (size_t)((b*16 + m)*64 + o);
      if (nt == 0) {
        size_t oi = row*64 + h*16 + n;
        Sout[oi] = acc[r] + (resid ? Sin[oi] : 0.f);
      } else {
        int cc = (nt - 1)*16 + n;
        if (cc < 24) {
          size_t oi = row*96 + (h*8 + cc/3)*3 + cc%3;
          Vout[oi] = acc[r] + (resid ? Vin[oi] : 0.f);
        }
      }
    }
  }
}

// ---------------------------------------------------------------------------
// Kernel 7: orbital attention. Block per (b,n,h); 4 waves share the 64x64 head.
// ---------------------------------------------------------------------------
__global__ void k_orb_attn(const float* __restrict__ QS, const float* __restrict__ KS,
                           const float* __restrict__ VSC,
                           const float* __restrict__ QV, const float* __restrict__ KV,
                           const float* __restrict__ VVE,
                           const float* __restrict__ Sin, const float* __restrict__ Vin,
                           float* __restrict__ Sout, float* __restrict__ Vout) {
  __shared__ __align__(16) _Float16 qa[64*64];   // [i][k]
  __shared__ __align__(16) _Float16 ka[64*64];   // [j][k]
  __shared__ __align__(16) _Float16 va[48*64];   // [c][j]
  __shared__ __align__(16) float    lg[64*64];   // [j][i]
  __shared__ __align__(16) _Float16 wt[64*64];   // [i][j]
  int blk = blockIdx.x;
  int h = blk & 3;
  int nn_ = (blk >> 2) & 15;
  int b = blk >> 6;
  int base = (b*16 + nn_)*64;
  const float ANORM  = 0.20412414523193154f;
  const float ANORM3 = ANORM * 0.57735026918962576f;
  for (int idx = threadIdx.x; idx < 64*64; idx += 128) {
    int i = idx >> 6, c = idx & 63;
    size_t row = (size_t)(base + i);
    float fq, fk;
    if (c < 16) {
      size_t oi = row*64 + h*16 + c;
      fq = QS[oi] * ANORM; fk = KS[oi];
    } else if (c < 40) {
      int cc = c - 16;
      size_t oi = row*96 + (h*8 + cc/3)*3 + cc%3;
      fq = QV[oi] * ANORM3; fk = KV[oi];
    } else { fq = fk = 0.f; }
    qa[idx] = (_Float16)fq;
    ka[idx] = (_Float16)fk;
  }
  for (int idx = threadIdx.x; idx < 48*64; idx += 128) {  // V, N-major [c][j]
    int c = idx >> 6, j = idx & 63;
    size_t row = (size_t)(base + j);
    float fv;
    if (c < 16)      fv = VSC[row*64 + h*16 + c];
    else if (c < 40) { int cc = c - 16; fv = VVE[row*96 + (h*8 + cc/3)*3 + cc%3]; }
    else             fv = 0.f;
    va[idx] = (_Float16)fv;
  }
  __syncthreads();
  int wv = threadIdx.x >> 5, lane = threadIdx.x & 31;
  int m0 = wv*16, n = lane & 15, hi = lane >> 4;
  for (int it = 0; it < 4; ++it) {                 // logits^T tiles
    v8f acc = {};
    acc = mmacc(ka, 64, qa, 64, m0, it*16, 64, acc);
#pragma unroll
    for (int r = 0; r < 8; ++r)
      lg[(m0 + r + 8*hi)*64 + it*16 + n] = acc[r];
  }
  __syncthreads();
  if (threadIdx.x < 64) {
    int i = threadIdx.x;
    float mx = -1e30f;
    for (int j = 0; j < 64; ++j) mx = fmaxf(mx, lg[j*64 + i]);
    float sum = 0.f;
    for (int j = 0; j < 64; ++j) sum += __expf(lg[j*64 + i] - mx);
    float inv = 1.f / sum;
    for (int j = 0; j < 64; ++j)
      wt[i*64 + j] = (_Float16)(__expf(lg[j*64 + i] - mx) * inv);
  }
  __syncthreads();
  for (int nt = 0; nt < 3; ++nt) {
    v8f acc = {};
    acc = mmacc(wt, 64, va, 64, m0, nt*16, 64, acc);
#pragma unroll
    for (int r = 0; r < 8; ++r) {
      int m = m0 + r + 8*hi;
      size_t row = (size_t)(base + m);
      if (nt == 0) {
        size_t oi = row*64 + h*16 + n;
        Sout[oi] = acc[r] + Sin[oi];
      } else {
        int cc = (nt - 1)*16 + n;
        if (cc < 24) {
          size_t oi = row*96 + (h*8 + cc/3)*3 + cc%3;
          Vout[oi] = acc[r] + Vin[oi];
        }
      }
    }
  }
}

// ---------------------------------------------------------------------------
// Kernel 8: gather occupied orbitals -> output (B,N,32,160)
// ---------------------------------------------------------------------------
__global__ void k_final(const float* __restrict__ S, const float* __restrict__ V,
                        float* __restrict__ out) {
  const int TOT = B_*N_*32*160;
  int tid = blockIdx.x*256 + threadIdx.x;
  if (tid >= TOT) return;
  int c = tid % 160;
  int r = tid / 160;
  int o = r % 32;
  int bn = r / 32;
  size_t row = (size_t)(bn*64 + o);
  out[tid] = (c < 64) ? S[row*64 + c] : V[row*96 + (c - 64)];
}

// ---------------------------------------------------------------------------
extern "C" void kernel_launch(void* const* d_in, const int* in_sizes, int n_in,
                              void* d_out, int out_size, void* d_ws, size_t ws_size,
                              hipStream_t stream) {
  (void)in_sizes; (void)n_in; (void)out_size; (void)ws_size;
  const float* scal = (const float*)d_in[0];
  const float* vec  = (const float*)d_in[1];
  const float* diff = (const float*)d_in[2];
  const float* dist = (const float*)d_in[3];
  const int*   nocc = (const int*)d_in[4];
  PPtrs P;
  for (int i = 0; i < 36; ++i) P.p[i] = (const float*)d_in[5 + i];

  char* ws = (char*)d_ws;
  size_t off = 0;
  auto alloc = [&](size_t bytes) {
    size_t o = off; off += (bytes + 255) & ~(size_t)255; return o;
  };
  float* s_a = (float*)(ws + alloc((size_t)NROW*64*4));
  float* s_b = (float*)(ws + alloc((size_t)NROW*64*4));
  float* v_a = (float*)(ws + alloc((size_t)NROW*96*4));
  float* v_b = (float*)(ws + alloc((size_t)NROW*96*4));
  float* ea  = (float*)(ws + alloc((size_t)4096*32*4));
  float* y1  = (float*)(ws + alloc((size_t)4096*3*4));
  float* q_s = (float*)(ws + alloc((size_t)3*NROW*64*4));   // q|k|v scalar
  float* q_v = (float*)(ws + alloc((size_t)3*NROW*96*4));   // q|k|v vector
  float* ew  = (float*)(ws + alloc((size_t)4096*64*4*4));
  _Float16* wf = (_Float16*)(ws + alloc((size_t)100352*2));

  {
    int tot = NROW*64 + NROW*96 + 4096*32 + 4096*3;
    k_prep_state<<<(tot + 255)/256, 256, 0, stream>>>(scal, vec, diff, dist, nocc,
                                                      s_a, v_a, ea, y1);
  }
  k_prep_params<<<(100352 + 255)/256, 256, 0, stream>>>(P, wf);

  const int ATTN_H = 15360, MLP_H = 19456;
  float* ks_ = q_s + (size_t)NROW*64;
  float* vs_ = q_s + (size_t)2*NROW*64;
  float* kv_ = q_v + (size_t)NROW*96;
  float* vv_ = q_v + (size_t)2*NROW*96;

  float* sc = s_a; float* vc = v_a;
  float* sn = s_b; float* vn = v_b;
  for (int it = 0; it < 2; ++it) {
    // ---- ion attention ----
    const _Float16* wa = wf + (size_t)it*ATTN_H;
    const _Float16* wm = wf + (size_t)4*ATTN_H + (size_t)it*MLP_H;
    k_proj_s<<<dim3(NROW/64, 1, 3), 128, 0, stream>>>(sc, wa, q_s);
    k_proj_v<<<dim3((NROW*96 + 255)/256, 1, 3), 256, 0, stream>>>(vc, wa, q_v);
    k_edge_mlp<<<4096, 128, 0, stream>>>(sc, vc, ea, y1, wm,
        P.p[it*12 + 9], P.p[it*12 + 10], P.p[it*12 + 11], ew);
    k_ion_attn<<<B_*O_, 128, 0, stream>>>(q_s, ks_, vs_, q_v, kv_, vv_, ew,
        sc, vc, sn, vn, (it == 0) ? 0 : 1);
    { float* t = sc; sc = sn; sn = t; t = vc; vc = vn; vn = t; }
    // ---- orbital attention ----
    const _Float16* wo = wf + (size_t)(2 + it)*ATTN_H;
    k_proj_s<<<dim3(NROW/64, 1, 3), 128, 0, stream>>>(sc, wo, q_s);
    k_proj_v<<<dim3((NROW*96 + 255)/256, 1, 3), 256, 0, stream>>>(vc, wo, q_v);
    k_orb_attn<<<B_*N_*H_, 128, 0, stream>>>(q_s, ks_, vs_, q_v, kv_, vv_,
                                             sc, vc, sn, vn);
    { float* t = sc; sc = sn; sn = t; t = vc; vc = vn; vn = t; }
  }
  k_final<<<(B_*N_*32*160 + 255)/256, 256, 0, stream>>>(sc, vc, (float*)d_out);
}
